// SparseNetwork2_1460288880653
// MI455X (gfx1250) — compile-verified, matrix operand order checked by reading, not run on hardware
//
#include <hip/hip_runtime.h>

// ---------------------------------------------------------------------------
// SparseNetwork2 on MI455X (gfx1250):
//   Densify the fixed-FAN_IN edge lists into dense f16 weight matrices each
//   call, then run the network as f16 WMMA GEMMs (f32 accumulate).
//   GEMM staging uses CDNA5 GLOBAL_LOAD_ASYNC_TO_LDS_B128 (ASYNCcnt) with
//   double-buffered LDS so the next K-tile copies while WMMA consumes the
//   current one. Activations kept batch-major [512][dim] f16.
// ---------------------------------------------------------------------------

typedef __attribute__((ext_vector_type(16))) _Float16 v16h;
typedef __attribute__((ext_vector_type(8)))  _Float16 v8h;
typedef __attribute__((ext_vector_type(8)))  float    v8f;

#define FAN_IN   128
#define BATCH    512
#define N_ROWS   3586          // all non-input nodes (incl. output layer)

// workspace layout (bytes)
#define OFF_W1   0u                     // 2048x4096 f16 = 16777216
#define OFF_W2   16777216u              // 1024x2048 f16 =  4194304
#define OFF_W3   20971520u              //  512x1024 f16 =  1048576
#define OFF_W4   22020096u              //    2x512  f16 =     2048
#define OFF_XH   22022144u              // 512x4096 f16  =  4194304
#define OFF_A1   26216448u              // 512x2048 f16  =  2097152
#define OFF_A2   28313600u              // 512x1024 f16  =  1048576
#define OFF_A3   29362176u              // 512x512  f16  =   524288
#define OFF_PRE  29886464u              // 512x2048 f32  =  4194304 (reused)
#define W_BYTES  22022144u              // dense-weight region to zero

// -------- CDNA5 async global->LDS copy (no VGPR round-trip, ASYNCcnt) ------
// Flat pointers to LDS carry the wave-relative LDS offset in their low 32
// bits (ISA 10.2 aperture mapping), which is exactly what VDST needs.
__device__ __forceinline__ void async_copy_b128(void* lds_dst, const void* gsrc) {
    unsigned lds_off = (unsigned)(size_t)lds_dst;
    asm volatile("global_load_async_to_lds_b128 %0, %1, off"
                 :: "v"(lds_off), "v"(gsrc) : "memory");
}
__device__ __forceinline__ void wait_async0() {
    asm volatile("s_wait_asynccnt 0x0" ::: "memory");
}

// ---------------------------------------------------------------- zero -----
__global__ void zero_kernel(uint4* __restrict__ p, int n16) {
    for (int i = blockIdx.x * blockDim.x + threadIdx.x; i < n16;
         i += gridDim.x * blockDim.x)
        p[i] = uint4{0u, 0u, 0u, 0u};
}

// ------------------------------------------------------------- convert -----
__global__ void cvt_f32_f16(const float* __restrict__ x,
                            _Float16* __restrict__ xh, int n) {
    for (int i = blockIdx.x * blockDim.x + threadIdx.x; i < n;
         i += gridDim.x * blockDim.x)
        xh[i] = (_Float16)x[i];
}

// ------------------------------------------------------------- scatter -----
// One thread per target row: its 128 edges are contiguous (edge base = r*128).
// Sequential per-row read-modify-write handles duplicate (src,tgt) edges
// exactly like the reference segment_sum, with no atomics needed.
__global__ void scatter_kernel(const float* __restrict__ weight,
                               const int*   __restrict__ edge_src,
                               _Float16* __restrict__ W1, _Float16* __restrict__ W2,
                               _Float16* __restrict__ W3, _Float16* __restrict__ W4) {
    int r = blockIdx.x * blockDim.x + threadIdx.x;
    if (r >= N_ROWS) return;
    _Float16* Wp; int Kd, prev, tl;
    if (r < 2048)      { Wp = W1; Kd = 4096; prev = 0;    tl = r;        }
    else if (r < 3072) { Wp = W2; Kd = 2048; prev = 4096; tl = r - 2048; }
    else if (r < 3584) { Wp = W3; Kd = 1024; prev = 6144; tl = r - 3072; }
    else               { Wp = W4; Kd = 512;  prev = 7168; tl = r - 3584; }
    const int base = r * FAN_IN;
    _Float16* row = Wp + (size_t)tl * Kd;
    for (int k = 0; k < FAN_IN; ++k) {
        int s = edge_src[base + k] - prev;
        row[s] = (_Float16)((float)row[s] + weight[base + k]);
    }
}

// ---------------------------------------------------------------- GEMM -----
// pre[b][t] = bias[boff+t] + sum_k W[t][k] * act[b][k]
// 256 threads = 8 waves; macro-tile 128(M) x 64(N); wave tile 32x32
// (2x2 v_wmma_f32_16x16x32_f16); K stepped by 32 through double-buffered LDS
// filled by global_load_async_to_lds_b128.
__launch_bounds__(256)
__global__ void gemm_wmma(const _Float16* __restrict__ W,    // [M][K]
                          const _Float16* __restrict__ act,  // [BATCH][K]
                          const float*    __restrict__ bias,
                          float*          __restrict__ pre,  // [BATCH][M]
                          int M, int K, int boff) {
    __shared__ __align__(16) _Float16 As[2][128 * 32];  // A tiles  (M x K)
    __shared__ __align__(16) _Float16 Bs[2][64 * 32];   // B^T tiles (N x K)

    const int tid  = threadIdx.x;
    const int lane = tid & 31;
    const int wid  = tid >> 5;
    const int wm   = wid & 3;            // wave M index (0..3)
    const int wn   = wid >> 2;           // wave N index (0..1)
    const int t0   = blockIdx.x * 128;
    const int n0   = blockIdx.y * 64;
    const int half = lane >> 4;          // ISA A/B frag: K-half select
    const int lrow = lane & 15;          // row (A) / col (B) within 16-tile
    const int klo  = half * 8;

    // per-thread staging coordinates (uint4 granularity)
    const int ar0 = tid >> 2,        ac = (tid & 3) * 8;   // A chunk 0: rows 0..63
    const int ar1 = (tid + 256) >> 2;                      // A chunk 1: rows 64..127
    const int br  = tid >> 2;                              // B rows 0..63

    auto stage = [&](int buf, int kk) {
        async_copy_b128(&As[buf][ar0 * 32 + ac], &W[(size_t)(t0 + ar0) * K + kk + ac]);
        async_copy_b128(&As[buf][ar1 * 32 + ac], &W[(size_t)(t0 + ar1) * K + kk + ac]);
        async_copy_b128(&Bs[buf][br  * 32 + ac], &act[(size_t)(n0 + br) * K + kk + ac]);
    };

    v8f c00 = {}, c01 = {}, c10 = {}, c11 = {};

    const int nk = K >> 5;
    stage(0, 0);
    wait_async0();
    __syncthreads();

    for (int ik = 0; ik < nk; ++ik) {
        const int buf = ik & 1;
        if (ik + 1 < nk) stage(buf ^ 1, (ik + 1) << 5);   // overlaps with compute

        // fragment loads per ISA layout:
        //   lane<16 : K in {0..7, 16..23};  lane>=16 : K in {8..15, 24..31}
        v16h a0, a1, b0, b1;
        {
            const _Float16* p = &As[buf][(wm * 32 + lrow) * 32 + klo];
            v8h lo = *(const v8h*)p, hi = *(const v8h*)(p + 16);
            a0 = __builtin_shufflevector(lo, hi, 0,1,2,3,4,5,6,7,8,9,10,11,12,13,14,15);
        }
        {
            const _Float16* p = &As[buf][(wm * 32 + 16 + lrow) * 32 + klo];
            v8h lo = *(const v8h*)p, hi = *(const v8h*)(p + 16);
            a1 = __builtin_shufflevector(lo, hi, 0,1,2,3,4,5,6,7,8,9,10,11,12,13,14,15);
        }
        {
            const _Float16* p = &Bs[buf][(wn * 32 + lrow) * 32 + klo];
            v8h lo = *(const v8h*)p, hi = *(const v8h*)(p + 16);
            b0 = __builtin_shufflevector(lo, hi, 0,1,2,3,4,5,6,7,8,9,10,11,12,13,14,15);
        }
        {
            const _Float16* p = &Bs[buf][(wn * 32 + 16 + lrow) * 32 + klo];
            v8h lo = *(const v8h*)p, hi = *(const v8h*)(p + 16);
            b1 = __builtin_shufflevector(lo, hi, 0,1,2,3,4,5,6,7,8,9,10,11,12,13,14,15);
        }

        c00 = __builtin_amdgcn_wmma_f32_16x16x32_f16(false, a0, false, b0, (short)0, c00, false, false);
        c01 = __builtin_amdgcn_wmma_f32_16x16x32_f16(false, a0, false, b1, (short)0, c01, false, false);
        c10 = __builtin_amdgcn_wmma_f32_16x16x32_f16(false, a1, false, b0, (short)0, c10, false, false);
        c11 = __builtin_amdgcn_wmma_f32_16x16x32_f16(false, a1, false, b1, (short)0, c11, false, false);

        wait_async0();      // own async copies (next buffer) done
        __syncthreads();    // everyone done reading this buffer / writing next
    }

    // epilogue: C layout -> lane L: N = L&15, M = 8*(L>>4) + j (j = VGPR idx).
    // pre is batch-major, so each lane writes 8 contiguous floats + bias.
    const v8f* cs[4] = { &c00, &c01, &c10, &c11 };
#pragma unroll
    for (int q = 0; q < 4; ++q) {
        int mi = q >> 1, ni = q & 1;
        int t = t0 + wm * 32 + mi * 16 + half * 8;
        int b = n0 + wn * 32 + ni * 16 + lrow;
        float*        dst = &pre[(size_t)b * M + t];
        const float4* bs  = (const float4*)&bias[boff + t];
        const v8f     cv  = *cs[q];
        float4 o0, o1;
        float4 b0v = bs[0], b1v = bs[1];
        o0.x = cv[0] + b0v.x; o0.y = cv[1] + b0v.y; o0.z = cv[2] + b0v.z; o0.w = cv[3] + b0v.w;
        o1.x = cv[4] + b1v.x; o1.y = cv[5] + b1v.y; o1.z = cv[6] + b1v.z; o1.w = cv[7] + b1v.w;
        ((float4*)dst)[0] = o0;
        ((float4*)dst)[1] = o1;
    }
}

// ------------------------------------------------------------ LN + ReLU ----
// One wave per batch sample: contiguous row reduction in pre[b][*].
__launch_bounds__(256)
__global__ void ln_relu_kernel(const float* __restrict__ pre,
                               const float* __restrict__ gamma,
                               const float* __restrict__ beta,
                               _Float16* __restrict__ act,
                               int dim, int goff) {
    const int wid = threadIdx.x >> 5, lane = threadIdx.x & 31;
    const int b = blockIdx.x * 8 + wid;
    const float* row = pre + (size_t)b * dim;
    float s = 0.f, s2 = 0.f;
    for (int t = lane; t < dim; t += 32) { float v = row[t]; s += v; s2 += v * v; }
    for (int off = 16; off > 0; off >>= 1) {
        s  += __shfl_xor(s,  off, 32);
        s2 += __shfl_xor(s2, off, 32);
    }
    const float inv_n = 1.0f / (float)dim;
    const float mean  = s * inv_n;
    const float var   = fmaxf(s2 * inv_n - mean * mean, 0.0f);
    const float rstd  = __frsqrt_rn(var + 1e-5f);
    _Float16* arow = act + (size_t)b * dim;
    for (int t = lane; t < dim; t += 32) {
        float v = (row[t] - mean) * rstd * gamma[goff + t] + beta[goff + t];
        arow[t] = (_Float16)fmaxf(v, 0.0f);
    }
}

// ------------------------------------------------------------ final layer --
// out[b][o] = bias4[o] + sum_s W4[o][s] * a3[b][s]   (2 outputs, K=512)
__launch_bounds__(256)
__global__ void final_kernel(const _Float16* __restrict__ W4,
                             const _Float16* __restrict__ a3,
                             const float* __restrict__ bias,
                             float* __restrict__ out) {
    const int wid = threadIdx.x >> 5, lane = threadIdx.x & 31;
    const int b = blockIdx.x * 8 + wid;
    const _Float16* row = a3 + (size_t)b * 512;
    float p0 = 0.f, p1 = 0.f;
    for (int s = lane; s < 512; s += 32) {
        float a = (float)row[s];
        p0 += a * (float)W4[s];
        p1 += a * (float)W4[512 + s];
    }
    for (int off = 16; off > 0; off >>= 1) {
        p0 += __shfl_xor(p0, off, 32);
        p1 += __shfl_xor(p1, off, 32);
    }
    if (lane == 0) {
        out[b * 2 + 0] = p0 + bias[3584];
        out[b * 2 + 1] = p1 + bias[3585];
    }
}

// ---------------------------------------------------------------------------
extern "C" void kernel_launch(void* const* d_in, const int* in_sizes, int n_in,
                              void* d_out, int out_size, void* d_ws, size_t ws_size,
                              hipStream_t stream) {
    const float* x      = (const float*)d_in[0];   // [512][4096]
    const float* weight = (const float*)d_in[1];   // [459008]
    const float* bias   = (const float*)d_in[2];   // [3586]
    const float* gamma  = (const float*)d_in[3];   // [3584]
    const float* beta   = (const float*)d_in[4];   // [3584]
    const int*   esrc   = (const int*)d_in[5];     // [459008]
    (void)in_sizes; (void)n_in; (void)out_size; (void)ws_size;

    char* ws = (char*)d_ws;
    _Float16* W1  = (_Float16*)(ws + OFF_W1);
    _Float16* W2  = (_Float16*)(ws + OFF_W2);
    _Float16* W3  = (_Float16*)(ws + OFF_W3);
    _Float16* W4  = (_Float16*)(ws + OFF_W4);
    _Float16* xh  = (_Float16*)(ws + OFF_XH);
    _Float16* a1  = (_Float16*)(ws + OFF_A1);
    _Float16* a2  = (_Float16*)(ws + OFF_A2);
    _Float16* a3  = (_Float16*)(ws + OFF_A3);
    float*    pre = (float*)   (ws + OFF_PRE);

    // 1) densify weights
    zero_kernel<<<512, 256, 0, stream>>>((uint4*)ws, (int)(W_BYTES / 16));
    scatter_kernel<<<(N_ROWS + 255) / 256, 256, 0, stream>>>(weight, esrc, W1, W2, W3, W4);

    // 2) convert input to f16 (already batch-major)
    cvt_f32_f16<<<1024, 256, 0, stream>>>(x, xh, BATCH * 4096);

    // 3) layer 1: 2048 <- 4096
    gemm_wmma<<<dim3(16, 8), 256, 0, stream>>>(W1, xh, bias, pre, 2048, 4096, 0);
    ln_relu_kernel<<<64, 256, 0, stream>>>(pre, gamma, beta, a1, 2048, 0);

    // 4) layer 2: 1024 <- 2048
    gemm_wmma<<<dim3(8, 8), 256, 0, stream>>>(W2, a1, bias, pre, 1024, 2048, 2048);
    ln_relu_kernel<<<64, 256, 0, stream>>>(pre, gamma, beta, a2, 1024, 2048);

    // 5) layer 3: 512 <- 1024
    gemm_wmma<<<dim3(4, 8), 256, 0, stream>>>(W3, a2, bias, pre, 512, 1024, 3072);
    ln_relu_kernel<<<64, 256, 0, stream>>>(pre, gamma, beta, a3, 512, 3072);

    // 6) output layer: 2 <- 512
    final_kernel<<<64, 256, 0, stream>>>(W4, a3, bias, (float*)d_out);
}
